// STGAT_encoder_16071767621793
// MI455X (gfx1250) — compile-verified, matrix operand order checked by reading, not run on hardware
//
#include <hip/hip_runtime.h>
#include <hip/hip_bf16.h>

#define TT 8
#define SS 512
#define PP 64
#define NN (SS * PP) // 32768

typedef __attribute__((ext_vector_type(16))) _Float16 v16h;
typedef __attribute__((ext_vector_type(8)))  float    v8f;

// D = A(16x32 f16) * B(32x16 f16) + C(16x16 f32)
__device__ __forceinline__ v8f wmma16x16x32(v16h a, v16h b, v8f c) {
  return __builtin_amdgcn_wmma_f32_16x16x32_f16(false, a, false, b, (short)0, c,
                                                false, false);
}

// ---- fast transcendentals on the CDNA5 TRANS pipe (co-executes with WMMA) ----
#if __has_builtin(__builtin_amdgcn_exp2f)
__device__ __forceinline__ float fexp2(float x) { return __builtin_amdgcn_exp2f(x); }
#else
__device__ __forceinline__ float fexp2(float x) { return exp2f(x); }
#endif
#if __has_builtin(__builtin_amdgcn_rcpf)
__device__ __forceinline__ float frcp(float x) { return __builtin_amdgcn_rcpf(x); }
#else
__device__ __forceinline__ float frcp(float x) { return 1.f / x; }
#endif
#if __has_builtin(__builtin_amdgcn_rsqf)
__device__ __forceinline__ float frsq(float x) { return __builtin_amdgcn_rsqf(x); }
#else
__device__ __forceinline__ float frsq(float x) { return rsqrtf(x); }
#endif
__device__ __forceinline__ float fexp(float x) { return fexp2(x * 1.44269504f); }
__device__ __forceinline__ float fsigmoid(float x) { return frcp(1.f + fexp(-x)); }
#if __has_builtin(__builtin_amdgcn_tanhf)
__device__ __forceinline__ float ftanh(float x) { return __builtin_amdgcn_tanhf(x); }
#else
__device__ __forceinline__ float ftanh(float x) {
  return 1.f - 2.f * frcp(fexp(2.f * x) + 1.f);  // inf/0 endpoints -> +/-1
}
#endif

// K base for fragment element pair v (0..7), lane half hf (0/1):
// VGPR v<4 -> K = 2v + 8*hf ; v>=4 -> K = 16 + 2(v-4) + 8*hf   (ISA 7.12.2, 16-bit A/B)
#define KB(v, hf) ((((v) < 4) ? 0 : 16) + (((v) & 3) << 1) + ((hf) << 3))

// ---------------------------------------------------------------------------
// LSTM: each wave owns 16 rows; Whh^T / Wih^T B-fragments live in VGPRs for all
// T steps (launch_bounds(128,1) -> full VGPR budget, no scratch spills);
// h cycles through a per-wave LDS slab (C-layout -> A-layout).
// ---------------------------------------------------------------------------
__global__ __launch_bounds__(128, 1) void lstm_kernel(
    const float* __restrict__ x, int xdim,
    const float* __restrict__ h0, const float* __restrict__ c0,
    const float* __restrict__ Wih, const float* __restrict__ Whh,
    const float* __restrict__ bih, const float* __restrict__ bhh,
    float* __restrict__ hs /* [T][N][32] */)
{
  __shared__ float hstage[4][512];
  const int lane = threadIdx.x & 31;
  const int wave = threadIdx.x >> 5;
  const int hf = lane >> 4;
  const int li = lane & 15;
  const int rowBase = (blockIdx.x * 4 + wave) * 16;
  float* hst = hstage[wave];

  // Preload B fragments: B[k][n] = W[n][k]  (g = x @ W^T)
  v16h bWhh[8], bWih[8];
  float biasv[8];
  #pragma unroll
  for (int n0 = 0; n0 < 8; ++n0) {
    const int n = n0 * 16 + li;
    v16h bh, bx;
    #pragma unroll
    for (int v = 0; v < 8; ++v) {
      const int k = KB(v, hf);
      bh[2*v]   = (_Float16)Whh[n*32 + k];
      bh[2*v+1] = (_Float16)Whh[n*32 + k + 1];
      bx[2*v]   = (k     < xdim) ? (_Float16)Wih[n*xdim + k]     : (_Float16)0.f;
      bx[2*v+1] = (k + 1 < xdim) ? (_Float16)Wih[n*xdim + k + 1] : (_Float16)0.f;
    }
    bWhh[n0] = bh; bWih[n0] = bx;
    biasv[n0] = bih[n] + bhh[n];
  }

  // init h stage (row-major [16][32]) and c registers (C layout)
  for (int i = lane; i < 512; i += 32) hst[i] = h0[(size_t)rowBase * 32 + i];
  float creg[2][8];
  #pragma unroll
  for (int ct = 0; ct < 2; ++ct)
    #pragma unroll
    for (int r = 0; r < 8; ++r)
      creg[ct][r] = c0[(size_t)(rowBase + r + 8*hf)*32 + ct*16 + li];

  #pragma unroll 1
  for (int t = 0; t < TT; ++t) {
    // A fragments: h (16x32) and x (16x32, zero-padded past xdim)
    v16h ah, ax;
    const float* xr = x + ((size_t)t*NN + rowBase + li) * xdim;
    if (t + 1 < TT)  // prefetch next step's input slice (global_prefetch_b8)
      __builtin_prefetch(x + ((size_t)(t + 1)*NN + rowBase + li) * xdim, 0, 1);
    #pragma unroll
    for (int v = 0; v < 8; ++v) {
      const int k = KB(v, hf);
      ah[2*v]   = (_Float16)hst[li*32 + k];
      ah[2*v+1] = (_Float16)hst[li*32 + k + 1];
      ax[2*v]   = (k     < xdim) ? (_Float16)xr[k]     : (_Float16)0.f;
      ax[2*v+1] = (k + 1 < xdim) ? (_Float16)xr[k + 1] : (_Float16)0.f;
    }
    // process the two 16-col halves of h independently: each needs only gate
    // tiles {ct, ct+2, ct+4, ct+6} -> only 4 accumulators live at a time
    #pragma unroll
    for (int ct = 0; ct < 2; ++ct) {
      float g4[4][8];
      #pragma unroll
      for (int gi = 0; gi < 4; ++gi) {
        const int n0 = gi*2 + ct;  // gate tile (i/f/g/o for this col half)
        v8f acc;
        #pragma unroll
        for (int r = 0; r < 8; ++r) acc[r] = biasv[n0];
        acc = wmma16x16x32(ah, bWhh[n0], acc);
        acc = wmma16x16x32(ax, bWih[n0], acc);
        #pragma unroll
        for (int r = 0; r < 8; ++r) g4[gi][r] = acc[r];
      }
      #pragma unroll
      for (int r = 0; r < 8; ++r) {
        const float iv = g4[0][r];
        const float fv = g4[1][r];
        const float gv = g4[2][r];
        const float ov = g4[3][r];
        const float cn = fsigmoid(fv) * creg[ct][r] + fsigmoid(iv) * ftanh(gv);
        creg[ct][r] = cn;
        const float hn = fsigmoid(ov) * ftanh(cn);
        const int m = r + 8*hf;
        const int col = ct*16 + li;
        hst[m*32 + col] = hn;  // same-wave LDS: in-order, no barrier needed
        hs[((size_t)t*NN + rowBase + m)*32 + col] = hn;
      }
    }
  }
}

// ---------------------------------------------------------------------------
// Fused GAT stack: one block per scene-step (b = s*T + t), P=64 nodes in LDS.
// norm -> GAT0(4 heads, WMMA) -> elu -> norm -> GAT1(WMMA) -> graph_in
// ---------------------------------------------------------------------------
__global__ void gat_kernel(const float* __restrict__ traj_hs,
                           const float* __restrict__ w0, const float* __restrict__ a_src0,
                           const float* __restrict__ a_dst0, const float* __restrict__ bias0,
                           const float* __restrict__ w1, const float* __restrict__ a_src1,
                           const float* __restrict__ a_dst1, const float* __restrict__ bias1,
                           float* __restrict__ graph_in /* [T][N][32] */)
{
  __shared__ __align__(16) char smem[57344];
  __shared__ float stat[2][64];
  __shared__ float red[512];   // partial-reduction scratch

  float*     xf     = (float*)(smem);            // [64][32]   ph1
  _Float16*  xh     = (_Float16*)(smem + 8192);  // [64][32]   ph1-2
  float*     hp0    = (float*)(smem + 12288);    // [4][64][16] ph2-4
  float*     sd     = (float*)(smem + 28672);    // [2][4][64] ph3-4
  _Float16*  attnh  = (_Float16*)(smem + 30720); // [64][64]   ph4 (per head)
  float*     x2f    = (float*)(smem + 40960);    // [64][64]   ph4-5
  _Float16*  x2h    = (_Float16*)(smem);         // [64][64]   ph5-8
  float*     hp1    = (float*)(smem + 8192);     // [64][32]   ph6-8
  float*     sd1    = (float*)(smem + 16384);    // [2][64]    ph7
  _Float16*  attn1h = (_Float16*)(smem + 20480); // [64][64]   ph7-8

  const int tid  = threadIdx.x;
  const int lane = tid & 31;
  const int wave = tid >> 5;
  const int hf = lane >> 4;
  const int li = lane & 15;

  const int b = blockIdx.x;
  const int s = b / TT;
  const int t = b % TT;

  // ---- phase 1: load x tile, instance-norm over nodes -> f16
  const float* xin = traj_hs + ((size_t)t*NN + (size_t)s*PP) * 32;
  for (int i = tid; i < 64*32; i += 256) xf[i] = xin[i];
  __syncthreads();
  {
    const int f = tid & 31, q = tid >> 5;      // 8 partials per feature
    float sum = 0.f, sq = 0.f;
    #pragma unroll
    for (int j = 0; j < 8; ++j) { float v = xf[(q*8 + j)*32 + f]; sum += v; sq += v*v; }
    red[q*32 + f] = sum; red[256 + q*32 + f] = sq;
  }
  __syncthreads();
  if (tid < 32) {
    float sum = 0.f, sq = 0.f;
    #pragma unroll
    for (int q = 0; q < 8; ++q) { sum += red[q*32 + tid]; sq += red[256 + q*32 + tid]; }
    const float m = sum * (1.f/64.f);
    const float var = sq * (1.f/64.f) - m*m;
    stat[0][tid] = m;
    stat[1][tid] = frsq(var + 1e-5f);
  }
  __syncthreads();
  for (int i = tid; i < 64*32; i += 256) {
    const int f = i & 31;
    xh[i] = (_Float16)((xf[i] - stat[0][f]) * stat[1][f]);
  }
  __syncthreads();

  // ---- phase 2: hp0[h][n][o] = xn @ w0[h]  (16 tiles over 8 waves)
  for (int ttile = wave; ttile < 16; ttile += 8) {
    const int h = ttile >> 2, mt = ttile & 3;
    v16h a, bw;
    #pragma unroll
    for (int v = 0; v < 8; ++v) {
      const int k = KB(v, hf);
      a[2*v]    = xh[(mt*16 + li)*32 + k];
      a[2*v+1]  = xh[(mt*16 + li)*32 + k + 1];
      bw[2*v]   = (_Float16)w0[(h*32 + k)*16 + li];
      bw[2*v+1] = (_Float16)w0[(h*32 + k + 1)*16 + li];
    }
    v8f acc;
    #pragma unroll
    for (int r = 0; r < 8; ++r) acc[r] = 0.f;
    acc = wmma16x16x32(a, bw, acc);
    #pragma unroll
    for (int r = 0; r < 8; ++r)
      hp0[(h*64 + mt*16 + r + 8*hf)*16 + li] = acc[r];
  }
  __syncthreads();

  // ---- phase 3: s/d logits per (head, node)
  {
    const int h = tid >> 6, n = tid & 63;
    float sv = 0.f, dv = 0.f;
    #pragma unroll
    for (int o = 0; o < 16; ++o) {
      const float hv = hp0[(h*64 + n)*16 + o];
      sv += hv * a_src0[h*16 + o];
      dv += hv * a_dst0[h*16 + o];
    }
    sd[h*64 + n]       = sv;
    sd[256 + h*64 + n] = dv;
  }
  __syncthreads();

  // ---- phase 4: per head: leaky-relu softmax rows (4 threads/row),
  //               out0 = attn@hp0 (+bias, elu) -> x2f
  for (int h = 0; h < 4; ++h) {
    const int q = tid >> 6, n = tid & 63;      // quarter, row
    const float sn = sd[h*64 + n];
    const float* dn = sd + 256 + h*64;
    float pm = -3.4e38f;
    #pragma unroll
    for (int j = 0; j < 16; ++j) {
      float v = sn + dn[q*16 + j];
      v = (v > 0.f) ? v : 0.2f * v;
      pm = fmaxf(pm, v);
    }
    red[q*64 + n] = pm;
    __syncthreads();
    const float mx = fmaxf(fmaxf(red[n], red[64 + n]), fmaxf(red[128 + n], red[192 + n]));
    __syncthreads();                           // all maxes read before reuse
    float ps = 0.f;
    #pragma unroll
    for (int j = 0; j < 16; ++j) {
      float v = sn + dn[q*16 + j];
      v = (v > 0.f) ? v : 0.2f * v;
      const float e = fexp(v - mx);            // e <= 1, f16-safe
      attnh[n*64 + q*16 + j] = (_Float16)e;    // unnormalized
      ps += e;
    }
    red[q*64 + n] = ps;
    __syncthreads();
    const float inv = frcp(red[n] + red[64 + n] + red[128 + n] + red[192 + n]);
    #pragma unroll
    for (int j = 0; j < 16; ++j) {
      const int idx = n*64 + q*16 + j;
      attnh[idx] = (_Float16)((float)attnh[idx] * inv);
    }
    __syncthreads();
    if (wave < 4) {                 // wave-uniform branch: EXEC all-ones for WMMA
      const int mt = wave;
      v16h a0, a1, b0, b1;
      #pragma unroll
      for (int v = 0; v < 8; ++v) {
        const int k = KB(v, hf);
        a0[2*v]   = attnh[(mt*16 + li)*64 + k];
        a0[2*v+1] = attnh[(mt*16 + li)*64 + k + 1];
        a1[2*v]   = attnh[(mt*16 + li)*64 + 32 + k];
        a1[2*v+1] = attnh[(mt*16 + li)*64 + 32 + k + 1];
        b0[2*v]   = (_Float16)hp0[(h*64 + k)*16 + li];
        b0[2*v+1] = (_Float16)hp0[(h*64 + k + 1)*16 + li];
        b1[2*v]   = (_Float16)hp0[(h*64 + 32 + k)*16 + li];
        b1[2*v+1] = (_Float16)hp0[(h*64 + 32 + k + 1)*16 + li];
      }
      v8f acc;
      #pragma unroll
      for (int r = 0; r < 8; ++r) acc[r] = 0.f;
      acc = wmma16x16x32(a0, b0, acc);
      acc = wmma16x16x32(a1, b1, acc);
      #pragma unroll
      for (int r = 0; r < 8; ++r) {
        float v = acc[r] + bias0[li];
        v = (v > 0.f) ? v : (fexp(v) - 1.f);            // elu
        x2f[(mt*16 + r + 8*hf)*64 + h*16 + li] = v;     // transpose(0,2,1,3) fold
      }
    }
    __syncthreads();
  }

  // ---- phase 5: instance-norm x2 -> f16
  {
    const int f = tid & 63, q = tid >> 6;      // 4 partials per feature
    float sum = 0.f, sq = 0.f;
    #pragma unroll
    for (int j = 0; j < 16; ++j) { float v = x2f[(q*16 + j)*64 + f]; sum += v; sq += v*v; }
    red[q*64 + f] = sum; red[256 + q*64 + f] = sq;
  }
  __syncthreads();
  if (tid < 64) {
    float sum = 0.f, sq = 0.f;
    #pragma unroll
    for (int q = 0; q < 4; ++q) { sum += red[q*64 + tid]; sq += red[256 + q*64 + tid]; }
    const float m = sum * (1.f/64.f);
    const float var = sq * (1.f/64.f) - m*m;
    stat[0][tid] = m;
    stat[1][tid] = frsq(var + 1e-5f);
  }
  __syncthreads();
  for (int i = tid; i < 64*64; i += 256) {
    const int f = i & 63;
    x2h[i] = (_Float16)((x2f[i] - stat[0][f]) * stat[1][f]);
  }
  __syncthreads();

  // ---- phase 6: hp1 = x2n @ w1  (8 tiles, K=64 -> 2 WMMAs each)
  {
    const int mt = wave >> 1, nt = wave & 1;
    v16h a0, a1, b0, b1;
    #pragma unroll
    for (int v = 0; v < 8; ++v) {
      const int k = KB(v, hf);
      a0[2*v]   = x2h[(mt*16 + li)*64 + k];
      a0[2*v+1] = x2h[(mt*16 + li)*64 + k + 1];
      a1[2*v]   = x2h[(mt*16 + li)*64 + 32 + k];
      a1[2*v+1] = x2h[(mt*16 + li)*64 + 32 + k + 1];
      b0[2*v]   = (_Float16)w1[k*32 + nt*16 + li];
      b0[2*v+1] = (_Float16)w1[(k + 1)*32 + nt*16 + li];
      b1[2*v]   = (_Float16)w1[(k + 32)*32 + nt*16 + li];
      b1[2*v+1] = (_Float16)w1[(k + 33)*32 + nt*16 + li];
    }
    v8f acc;
    #pragma unroll
    for (int r = 0; r < 8; ++r) acc[r] = 0.f;
    acc = wmma16x16x32(a0, b0, acc);
    acc = wmma16x16x32(a1, b1, acc);
    #pragma unroll
    for (int r = 0; r < 8; ++r)
      hp1[(mt*16 + r + 8*hf)*32 + nt*16 + li] = acc[r];
  }
  __syncthreads();

  // ---- phase 7: layer-1 attention (4 threads/row softmax)
  if (tid < 64) {
    const int n = tid;
    float sv = 0.f, dv = 0.f;
    #pragma unroll
    for (int o = 0; o < 32; ++o) {
      const float hv = hp1[n*32 + o];
      sv += hv * a_src1[o];
      dv += hv * a_dst1[o];
    }
    sd1[n]      = sv;
    sd1[64 + n] = dv;
  }
  __syncthreads();
  {
    const int q = tid >> 6, n = tid & 63;
    const float sn = sd1[n];
    float pm = -3.4e38f;
    #pragma unroll
    for (int j = 0; j < 16; ++j) {
      float v = sn + sd1[64 + q*16 + j];
      v = (v > 0.f) ? v : 0.2f * v;
      pm = fmaxf(pm, v);
    }
    red[q*64 + n] = pm;
    __syncthreads();
    const float mx = fmaxf(fmaxf(red[n], red[64 + n]), fmaxf(red[128 + n], red[192 + n]));
    __syncthreads();
    float ps = 0.f;
    #pragma unroll
    for (int j = 0; j < 16; ++j) {
      float v = sn + sd1[64 + q*16 + j];
      v = (v > 0.f) ? v : 0.2f * v;
      const float e = fexp(v - mx);
      attn1h[n*64 + q*16 + j] = (_Float16)e;
      ps += e;
    }
    red[q*64 + n] = ps;
    __syncthreads();
    const float inv = frcp(red[n] + red[64 + n] + red[128 + n] + red[192 + n]);
    #pragma unroll
    for (int j = 0; j < 16; ++j) {
      const int idx = n*64 + q*16 + j;
      attn1h[idx] = (_Float16)((float)attn1h[idx] * inv);
    }
  }
  __syncthreads();

  // ---- phase 8: out1 = attn1 @ hp1 + bias1 -> graph_in[t][s*P+p][:]
  {
    const int mt = wave >> 1, nt = wave & 1;
    v16h a0, a1, b0, b1;
    #pragma unroll
    for (int v = 0; v < 8; ++v) {
      const int k = KB(v, hf);
      a0[2*v]   = attn1h[(mt*16 + li)*64 + k];
      a0[2*v+1] = attn1h[(mt*16 + li)*64 + k + 1];
      a1[2*v]   = attn1h[(mt*16 + li)*64 + 32 + k];
      a1[2*v+1] = attn1h[(mt*16 + li)*64 + 32 + k + 1];
      b0[2*v]   = (_Float16)hp1[k*32 + nt*16 + li];
      b0[2*v+1] = (_Float16)hp1[(k + 1)*32 + nt*16 + li];
      b1[2*v]   = (_Float16)hp1[(k + 32)*32 + nt*16 + li];
      b1[2*v+1] = (_Float16)hp1[(k + 33)*32 + nt*16 + li];
    }
    v8f acc;
    #pragma unroll
    for (int r = 0; r < 8; ++r) acc[r] = 0.f;
    acc = wmma16x16x32(a0, b0, acc);
    acc = wmma16x16x32(a1, b1, acc);
    #pragma unroll
    for (int r = 0; r < 8; ++r) {
      const int m = mt*16 + r + 8*hf;
      graph_in[((size_t)t*NN + (size_t)s*PP + m)*32 + nt*16 + li] =
          acc[r] + bias1[nt*16 + li];
    }
  }
}

// ---------------------------------------------------------------------------
// encoded = [traj_hs[-1] | graph_hs[-1] | repeat(z_noise, P)]
// ---------------------------------------------------------------------------
__global__ void concat_kernel(const float* __restrict__ traj_hs,
                              const float* __restrict__ graph_hs,
                              const float* __restrict__ z,
                              float* __restrict__ enc)
{
  const int i = blockIdx.x * blockDim.x + threadIdx.x;
  if (i >= NN * 72) return;
  const int n = i / 72, j = i % 72;
  float v;
  if (j < 32)      v = traj_hs[((size_t)(TT - 1)*NN + n)*32 + j];
  else if (j < 64) v = graph_hs[((size_t)(TT - 1)*NN + n)*32 + (j - 32)];
  else             v = z[(n / PP)*8 + (j - 64)];
  enc[i] = v;
}

extern "C" void kernel_launch(void* const* d_in, const int* in_sizes, int n_in,
                              void* d_out, int out_size, void* d_ws, size_t ws_size,
                              hipStream_t stream) {
  const float* obs      = (const float*)d_in[0];
  const float* h0_traj  = (const float*)d_in[1];
  const float* c0_traj  = (const float*)d_in[2];
  const float* h0_graph = (const float*)d_in[3];
  const float* c0_graph = (const float*)d_in[4];
  const float* z_noise  = (const float*)d_in[5];
  const float* Wih_t    = (const float*)d_in[6];
  const float* Whh_t    = (const float*)d_in[7];
  const float* bih_t    = (const float*)d_in[8];
  const float* bhh_t    = (const float*)d_in[9];
  const float* Wih_g    = (const float*)d_in[10];
  const float* Whh_g    = (const float*)d_in[11];
  const float* bih_g    = (const float*)d_in[12];
  const float* bhh_g    = (const float*)d_in[13];
  const float* w0       = (const float*)d_in[14];
  const float* a_src0   = (const float*)d_in[15];
  const float* a_dst0   = (const float*)d_in[16];
  const float* bias0    = (const float*)d_in[17];
  const float* w1       = (const float*)d_in[18];
  const float* a_src1   = (const float*)d_in[19];
  const float* a_dst1   = (const float*)d_in[20];
  const float* bias1    = (const float*)d_in[21];

  float* out      = (float*)d_out;
  float* enc      = out;                               // [N][72]
  float* graph_hs = out + (size_t)NN * 72;             // [T][N][32]
  float* traj_hs  = graph_hs + (size_t)TT * NN * 32;   // [T][N][32]
  float* graph_in = (float*)d_ws;                      // [T][N][32] scratch

  lstm_kernel<<<dim3(NN / 64), dim3(128), 0, stream>>>(
      obs, 3, h0_traj, c0_traj, Wih_t, Whh_t, bih_t, bhh_t, traj_hs);
  gat_kernel<<<dim3(SS * TT), dim3(256), 0, stream>>>(
      traj_hs, w0, a_src0, a_dst0, bias0, w1, a_src1, a_dst1, bias1, graph_in);
  lstm_kernel<<<dim3(NN / 64), dim3(128), 0, stream>>>(
      graph_in, 32, h0_graph, c0_graph, Wih_g, Whh_g, bih_g, bhh_g, graph_hs);
  concat_kernel<<<dim3((NN * 72 + 255) / 256), dim3(256), 0, stream>>>(
      traj_hs, graph_hs, z_noise, enc);

  (void)in_sizes; (void)n_in; (void)out_size; (void)ws_size;
}
